// SelfMultiHeadAttention_2422361555266
// MI455X (gfx1250) — matrix operand order repaired and checked by reading, not compiled, not run on hardware
//
#include <hip/hip_runtime.h>
#include <hip/hip_bf16.h>
#include <stdint.h>

#define BB 2
#define LL 2048
#define DD 1024
#define HH 16
#define HD 64

typedef __attribute__((ext_vector_type(16))) __bf16 v16bf;
typedef __attribute__((ext_vector_type(8)))  float  v8f;
typedef __attribute__((ext_vector_type(4)))  unsigned v4u;
typedef __attribute__((ext_vector_type(8)))  int      v8i;
typedef __attribute__((ext_vector_type(4)))  int      v4i;

union Frag { unsigned u[8]; v16bf v; };

__device__ __forceinline__ unsigned short f2bf(float f) {
  union { float f; unsigned u; } x; x.f = f;
  unsigned u = x.u;
  u += 0x7FFFu + ((u >> 16) & 1u);      // round-to-nearest-even
  return (unsigned short)(u >> 16);
}

__device__ __forceinline__ unsigned lds_off(const void* p) {
  return (unsigned)(unsigned long long)p;   // generic LDS addr low 32 bits = LDS byte offset
}

// ---- Tensor Data Mover: 2D bf16 tile (tile_w x tile_h) global -> LDS ----
__device__ __forceinline__ void tdm_load_2d_bf16(unsigned lds_byte, const void* gptr,
                                                 unsigned tile_w, unsigned tile_h,
                                                 unsigned row_stride_elems) {
  const unsigned long long ga = (unsigned long long)gptr;
  v4u g0;
  g0[0] = 1u;                                                 // count=1, user mode
  g0[1] = lds_byte;                                           // lds_addr
  g0[2] = (unsigned)ga;                                       // global_addr[31:0]
  g0[3] = ((unsigned)(ga >> 32) & 0x01FFFFFFu) | 0x80000000u; // addr[56:32] | type=2
  const unsigned td0 = row_stride_elems;                      // tensor_dim0 (elements)
  const unsigned td1 = 0x40000000u;                           // tensor_dim1: huge, no clip
  v8i g1;
  g1[0] = (int)(1u << 16);                                    // data_size=1 (2 bytes)
  g1[1] = (int)((td0 & 0xFFFFu) << 16);                       // tensor_dim0[15:0]
  g1[2] = (int)(((td0 >> 16) & 0xFFFFu) | ((td1 & 0xFFFFu) << 16));
  g1[3] = (int)(((td1 >> 16) & 0xFFFFu) | ((tile_w & 0xFFFFu) << 16)); // tile_dim0
  g1[4] = (int)(tile_h & 0xFFFFu);                            // tile_dim1, tile_dim2=0
  g1[5] = (int)row_stride_elems;                              // tensor_dim0_stride[31:0]
  g1[6] = 0;
  g1[7] = 0;
  const v4i gz = {0, 0, 0, 0};
#if __clang_major__ >= 23
  const v8i gz8 = {0, 0, 0, 0, 0, 0, 0, 0};
  __builtin_amdgcn_tensor_load_to_lds(g0, g1, gz, gz, gz8, 0);
#else
  __builtin_amdgcn_tensor_load_to_lds(g0, g1, gz, gz, 0);
#endif
}

// A fragment: 16x32 bf16 (ISA 7.12.2).
__device__ __forceinline__ void load_a_frag(const unsigned short* lds, int ld,
                                            int row0, int koff, int lane, Frag& a) {
  const int half = (lane >> 4) & 1;
  const unsigned short* r = lds + (row0 + (lane & 15)) * ld + koff + 16 * half;
#pragma unroll
  for (int j = 0; j < 8; ++j) {
    const int kb = 2 * j;
    a.u[j] = *(const unsigned*)(r + kb);
  }
}

// B fragment from LDS staged TRANSPOSED as [n][k].
__device__ __forceinline__ void load_b_frag(const unsigned short* lds, int ldk,
                                            int n0, int koff, int lane, Frag& b) {
  const int half = (lane >> 4) & 1;
  const unsigned short* r = lds + (n0 + (lane & 15)) * ldk + koff + 16 * half;
#pragma unroll
  for (int j = 0; j < 8; ++j) b.u[j] = *(const unsigned*)(r + 2 * j);
}

__device__ __forceinline__ v8f wmma_bf16(const Frag& a, const Frag& b, v8f c) {
  return __builtin_amdgcn_wmma_f32_16x16x32_bf16(false, a.v, false, b.v, (short)0, c,
                                                 false, false);
}

__global__ void __launch_bounds__(256) cvt_f32_bf16(const float* __restrict__ src,
                                                    unsigned short* __restrict__ dst, int n4) {
  const int i = blockIdx.x * blockDim.x + threadIdx.x;
  if (i < n4) {
    const float4 f = ((const float4*)src)[i];
    ushort4 o; o.x = f2bf(f.x); o.y = f2bf(f.y); o.z = f2bf(f.z); o.w = f2bf(f.w);
    ((ushort4*)dst)[i] = o;
  }
}

// Wt[n][k] (bf16) = W[k][n] (fp32); one-time LDS-tiled transpose+convert.
__global__ void __launch_bounds__(256)
cvt_transpose(const float* __restrict__ W, unsigned short* __restrict__ Wt, int K, int N) {
  __shared__ unsigned short s[32][33];
  const int k0 = blockIdx.y * 32, n0 = blockIdx.x * 32;
  const int row = threadIdx.x >> 5, col = threadIdx.x & 31;
#pragma unroll
  for (int i = 0; i < 4; ++i)
    s[row + 8 * i][col] = f2bf(W[(size_t)(k0 + row + 8 * i) * N + n0 + col]);
  __syncthreads();
#pragma unroll
  for (int i = 0; i < 4; ++i)
    Wt[(size_t)(n0 + row + 8 * i) * K + k0 + col] = s[col][row + 8 * i];
}

// C[4096,3072] = Xbf @ Wqkv + b. All-TDM double-buffered pipeline.
// Q,K written as [B*L][D] bf16; V written TRANSPOSED as [b][h][d][l].
__global__ void __launch_bounds__(256, 2)
qkv_gemm(const unsigned short* __restrict__ X, const unsigned short* __restrict__ Wt,
         const float* __restrict__ bias,
         unsigned short* __restrict__ Qb, unsigned short* __restrict__ Kb,
         unsigned short* __restrict__ Vt) {
  __shared__ unsigned short sA[2][128 * 32];
  __shared__ unsigned short sB[2][64 * 32];   // [n][k]
  const int m0 = blockIdx.y * 128, n0 = blockIdx.x * 64;
  const int t = threadIdx.x, lane = t & 31, w = t >> 5;
  const int wm = (w & 3) * 32, wn = (w >> 2) * 32;
  v8f c[2][2] = {};
  if (w == 0) {
    tdm_load_2d_bf16(lds_off(sA[0]), X + (size_t)m0 * DD, 32, 128, DD);
    tdm_load_2d_bf16(lds_off(sB[0]), Wt + (size_t)n0 * DD, 32, 64, DD);
    __builtin_amdgcn_s_wait_tensorcnt(0);
  }
  __syncthreads();
  for (int it = 0; it < DD / 32; ++it) {
    const int cur = it & 1, nxt = cur ^ 1;
    if (w == 0 && it + 1 < DD / 32) {
      const int k1 = (it + 1) * 32;
      tdm_load_2d_bf16(lds_off(sA[nxt]), X + (size_t)m0 * DD + k1, 32, 128, DD);
      tdm_load_2d_bf16(lds_off(sB[nxt]), Wt + (size_t)n0 * DD + k1, 32, 64, DD);
    }
    Frag a0, a1, b0, b1;
    load_a_frag(sA[cur], 32, wm, 0, lane, a0);
    load_a_frag(sA[cur], 32, wm + 16, 0, lane, a1);
    load_b_frag(sB[cur], 32, wn, 0, lane, b0);
    load_b_frag(sB[cur], 32, wn + 16, 0, lane, b1);
    c[0][0] = wmma_bf16(a0, b0, c[0][0]);
    c[0][1] = wmma_bf16(a0, b1, c[0][1]);
    c[1][0] = wmma_bf16(a1, b0, c[1][0]);
    c[1][1] = wmma_bf16(a1, b1, c[1][1]);
    if (w == 0) __builtin_amdgcn_s_wait_tensorcnt(0);
    __syncthreads();
  }
  const int half = (lane >> 4) & 1;
#pragma unroll
  for (int mi = 0; mi < 2; ++mi)
#pragma unroll
    for (int ni = 0; ni < 2; ++ni) {
      const int ncol = n0 + wn + ni * 16 + (lane & 15);
      const float bv = bias[ncol];
      if (ncol < 2 * DD) {                       // Q or K: [B*L][D]
        unsigned short* outp = (ncol < DD) ? Qb : Kb;
        const int cl = ncol & (DD - 1);
#pragma unroll
        for (int r = 0; r < 8; ++r) {
          const int m = m0 + wm + mi * 16 + r + 8 * half;
          outp[(size_t)m * DD + cl] = f2bf(c[mi][ni][r] + bv);
        }
      } else {                                   // V transposed: [b][h][d][l]
        const int hv = (ncol - 2 * DD) >> 6;
        const int dv = (ncol - 2 * DD) & 63;
#pragma unroll
        for (int r = 0; r < 8; ++r) {
          const int m = m0 + wm + mi * 16 + r + 8 * half;
          const int bb = m >> 11, l = m & (LL - 1);
          Vt[(((size_t)bb * HH + hv) * HD + dv) * LL + l] = f2bf(c[mi][ni][r] + bv);
        }
      }
    }
}

// Fused scores + softmax + AV. One block per (bh, 64-query tile).
// Pass 1: stream K tiles (TDM), WMMA scores, online per-row max / sum(exp).
// Pass 2: stream K+V tiles (TDM), recompute scores, write normalized attn ONCE,
//         convert tile to bf16 in LDS, WMMA-accumulate ctx.
__global__ void __launch_bounds__(256, 2)
attn_fused(const unsigned short* __restrict__ Qb, const unsigned short* __restrict__ Kb,
           const unsigned short* __restrict__ Vt, const float* __restrict__ mask,
           float* __restrict__ attn, unsigned short* __restrict__ Ctx) {
  __shared__ unsigned short sQ[64 * 64];
  __shared__ unsigned short sK[2][64 * 64];   // [n][k]
  __shared__ unsigned short sV[2][64 * 64];   // [d][k]
  __shared__ unsigned short sAt[64 * 64];     // bf16 attn tile [m][k]
  __shared__ float sS[64 * 64];
  __shared__ float pred[256];
  __shared__ float rowBM[64], rowM[64], rowS[64], rowInvS[64];

  const int m0 = blockIdx.x * 64, bh = blockIdx.y;
  const int b = bh >> 4, h = bh & 15;
  const unsigned short* Qh = Qb + (size_t)b * LL * DD + (size_t)h * LL * HD;
  const unsigned short* Kh = Kb + (size_t)b * LL * DD + (size_t)h * LL * HD;
  const unsigned short* Vh = Vt + (size_t)bh * HD * LL;       // [64][2048]
  unsigned short* Ch = Ctx + (size_t)b * LL * DD + (size_t)h * LL * HD;
  float* ap = attn + (size_t)bh * LL * LL;
  const int t = threadIdx.x, lane = t & 31, w = t >> 5;
  const int mt = (w & 3) * 16, nt0 = (w >> 2) * 32;
  const int half = (lane >> 4) & 1, ln = lane & 15;
  const int rrow = t >> 2, rq = (t & 3) * 16;

  if (t < 64) { rowM[t] = -3.0e38f; rowS[t] = 0.0f; }
  if (w == 0) {
    tdm_load_2d_bf16(lds_off(sQ), Qh + (size_t)m0 * HD, 64, 64, HD);
    tdm_load_2d_bf16(lds_off(sK[0]), Kh, 64, 64, HD);
    __builtin_amdgcn_s_wait_tensorcnt(0);
  }
  __syncthreads();

  // ---------------- pass 1 ----------------
  for (int it = 0; it < LL / 64; ++it) {
    const int cur = it & 1, nxt = cur ^ 1;
    if (w == 0 && it + 1 < LL / 64)
      tdm_load_2d_bf16(lds_off(sK[nxt]), Kh + (size_t)(it + 1) * 64 * HD, 64, 64, HD);
    v8f c[2] = {};
    Frag a0, a1, b0, b1;
    load_a_frag(sQ, 64, mt, 0, lane, a0);
    load_a_frag(sQ, 64, mt, 32, lane, a1);
    load_b_frag(sK[cur], 64, nt0, 0, lane, b0);
    load_b_frag(sK[cur], 64, nt0 + 16, 0, lane, b1);
    c[0] = wmma_bf16(a0, b0, c[0]);
    c[1] = wmma_bf16(a0, b1, c[1]);
    load_b_frag(sK[cur], 64, nt0, 32, lane, b0);
    load_b_frag(sK[cur], 64, nt0 + 16, 32, lane, b1);
    c[0] = wmma_bf16(a1, b0, c[0]);
    c[1] = wmma_bf16(a1, b1, c[1]);
    const int kbase = it * 64;
#pragma unroll
    for (int ni = 0; ni < 2; ++ni) {
      const int n = nt0 + ni * 16 + ln;
#pragma unroll
      for (int r = 0; r < 8; ++r) {
        const int m = mt + r + 8 * half;
        sS[m * 64 + n] = c[ni][r] * 0.03125f * mask[(size_t)(m0 + m) * LL + kbase + n];
      }
    }
    __syncthreads();
    float pm = sS[rrow * 64 + rq];
#pragma unroll
    for (int i = 1; i < 16; ++i) pm = fmaxf(pm, sS[rrow * 64 + rq + i]);
    pred[t] = pm;
    __syncthreads();
    if (t < 64)
      rowBM[t] = fmaxf(fmaxf(pred[t * 4], pred[t * 4 + 1]),
                       fmaxf(pred[t * 4 + 2], pred[t * 4 + 3]));
    __syncthreads();
    const float bm = rowBM[rrow];
    float ps = 0.0f;
#pragma unroll
    for (int i = 0; i < 16; ++i) ps += __expf(sS[rrow * 64 + rq + i] - bm);
    pred[t] = ps;
    if (w == 0) __builtin_amdgcn_s_wait_tensorcnt(0);
    __syncthreads();
    if (t < 64) {
      const float bs = pred[t * 4] + pred[t * 4 + 1] + pred[t * 4 + 2] + pred[t * 4 + 3];
      const float bmv = rowBM[t];
      const float mo = rowM[t];
      const float mn = fmaxf(mo, bmv);
      rowS[t] = rowS[t] * __expf(mo - mn) + bs * __expf(bmv - mn);
      rowM[t] = mn;
    }
  }
  __syncthreads();
  if (t < 64) rowInvS[t] = 1.0f / rowS[t];
  if (w == 0) {
    tdm_load_2d_bf16(lds_off(sK[0]), Kh, 64, 64, HD);
    tdm_load_2d_bf16(lds_off(sV[0]), Vh, 64, 64, LL);
    __builtin_amdgcn_s_wait_tensorcnt(0);
  }
  __syncthreads();

  // ---------------- pass 2 ----------------
  v8f ct[2] = {};
  for (int it = 0; it < LL / 64; ++it) {
    const int cur = it & 1, nxt = cur ^ 1;
    if (w == 0 && it + 1 < LL / 64) {
      tdm_load_2d_bf16(lds_off(sK[nxt]), Kh + (size_t)(it + 1) * 64 * HD, 64, 64, HD);
      tdm_load_2d_bf16(lds_off(sV[nxt]), Vh + (it + 1) * 64, 64, 64, LL);
    }
    v8f c[2] = {};
    Frag a0, a1, b0, b1;
    load_a_frag(sQ, 64, mt, 0, lane, a0);
    load_a_frag(sQ, 64, mt, 32, lane, a1);
    load_b_frag(sK[cur], 64, nt0, 0, lane, b0);
    load_b_frag(sK[cur], 64, nt0 + 16, 0, lane, b1);
    c[0] = wmma_bf16(a0, b0, c[0]);
    c[1] = wmma_bf16(a0, b1, c[1]);
    load_b_frag(sK[cur], 64, nt0, 32, lane, b0);
    load_b_frag(sK[cur], 64, nt0 + 16, 32, lane, b1);
    c[0] = wmma_bf16(a1, b0, c[0]);
    c[1] = wmma_bf16(a1, b1, c[1]);
    const int kbase = it * 64;
#pragma unroll
    for (int ni = 0; ni < 2; ++ni) {
      const int n = nt0 + ni * 16 + ln;
#pragma unroll
      for (int r = 0; r < 8; ++r) {
        const int m = mt + r + 8 * half;
        const float s = c[ni][r] * 0.03125f * mask[(size_t)(m0 + m) * LL + kbase + n];
        const float a = __expf(s - rowM[m]) * rowInvS[m];
        ap[(size_t)(m0 + m) * LL + kbase + n] = a;   // the only attn HBM write
        sAt[m * 64 + n] = f2bf(a);
      }
    }
    __syncthreads();
    Frag av0, av1, vb0, vb1;
    load_a_frag(sAt, 64, mt, 0, lane, av0);
    load_a_frag(sAt, 64, mt, 32, lane, av1);
    load_b_frag(sV[cur], 64, nt0, 0, lane, vb0);
    load_b_frag(sV[cur], 64, nt0 + 16, 0, lane, vb1);
    ct[0] = wmma_bf16(av0, vb0, ct[0]);
    ct[1] = wmma_bf16(av0, vb1, ct[1]);
    load_b_frag(sV[cur], 64, nt0, 32, lane, vb0);
    load_b_frag(sV[cur], 64, nt0 + 16, 32, lane, vb1);
    ct[0] = wmma_bf16(av1, vb0, ct[0]);
    ct[1] = wmma_bf16(av1, vb1, ct[1]);
    if (w == 0) __builtin_amdgcn_s_wait_tensorcnt(0);
    __syncthreads();
  }
#pragma unroll
  for (int ni = 0; ni < 2; ++ni) {
    const int d = nt0 + ni * 16 + ln;
#pragma unroll
    for (int r = 0; r < 8; ++r) {
      const int m = m0 + mt + r + 8 * half;
      Ch[(size_t)m * HD + d] = f2bf(ct[ni][r]);
    }
  }
}

// out = Ctx @ Wout + b_out (fp32). All-TDM double-buffered pipeline.
__global__ void __launch_bounds__(256, 2)
out_gemm(const unsigned short* __restrict__ Ctx, const unsigned short* __restrict__ Wt,
         const float* __restrict__ bias, float* __restrict__ out) {
  __shared__ unsigned short sA[2][128 * 32];
  __shared__ unsigned short sB[2][64 * 32];
  const int m0 = blockIdx.y * 128, n0 = blockIdx.x * 64;
  const int t = threadIdx.x, lane = t & 31, w = t >> 5;
  const int wm = (w & 3) * 32, wn = (w >> 2) * 32;
  v8f c[2][2] = {};
  if (w == 0) {
    tdm_load_2d_bf16(lds_off(sA[0]), Ctx + (size_t)m0 * DD, 32, 128, DD);
    tdm_load_2d_bf16(lds_off(sB[0]), Wt + (size_t)n0 * DD, 32, 64, DD);
    __builtin_amdgcn_s_wait_tensorcnt(0);
  }
  __syncthreads();
  for (int it = 0; it < DD / 32; ++it) {
    const int cur = it & 1, nxt = cur ^ 1;
    if (w == 0 && it + 1 < DD / 32) {
      const int k1 = (it + 1) * 32;
      tdm_load_2d_bf16(lds_off(sA[nxt]), Ctx + (size_t)m0 * DD + k1, 32, 128, DD);
      tdm_load_2d_bf16(lds_off(sB[nxt]), Wt + (size_t)n0 * DD + k1, 32, 64, DD);
    }
    Frag a0, a1, b0, b1;
    load_a_frag(sA[cur], 32, wm, 0, lane, a0);
    load_a_frag(sA[cur], 32, wm + 16, 0, lane, a1);
    load_b_frag(sB[cur], 32, wn, 0, lane, b0);
    load_b_frag(sB[cur], 32, wn + 16, 0, lane, b1);
    c[0][0] = wmma_bf16(a0, b0, c[0][0]);
    c[0][1] = wmma_bf16(a0, b1, c[0][1]);
    c[1][0] = wmma_bf16(a1, b0, c[1][0]);
    c[1][1] = wmma_bf16(a1, b1, c[1][1]);
    if (w == 0) __builtin_amdgcn_s_wait_tensorcnt(0);
    __syncthreads();
  }
  const int half = (lane >> 4) & 1;
#pragma unroll
  for (int mi = 0; mi < 2; ++mi)
#pragma unroll
    for (int ni = 0; ni < 2; ++ni) {
      const int n = n0 + wn + ni * 16 + (lane & 15);
      const float bv = bias[n];
#pragma unroll
      for (int r = 0; r < 8; ++r) {
        const int m = m0 + wm + mi * 16 + r + 8 * half;
        out[(size_t)m * DD + n] = c[mi][ni][r] + bv;
      }
    }
}

extern "C" void kernel_launch(void* const* d_in, const int* in_sizes, int n_in,
                              void* d_out, int out_size, void* d_ws, size_t ws_size,
                              hipStream_t stream) {
  (void)in_sizes; (void)n_in; (void)out_size; (void)ws_size;
  const float* x    = (const float*)d_in[0];
  const float* mask = (const float*)d_in[1];
  const float* Wqkv = (const float*)d_in[2];
  const float* bqkv = (const float*)d_in[3];
  const float* Wout = (const float*)d_in[4];
  const float* bout = (const float*)d_in[5];
  float* out  = (float*)d_out;
  float* attn = out + (size_t)BB * LL * DD;

  char* ws = (char*)d_ws;
  unsigned short* Xbf = (unsigned short*)ws; ws += (size_t)BB * LL * DD * 2;
  unsigned short* Wqt = (unsigned short*)ws; ws += (size_t)3 * DD * DD * 2;  // [3072][1024]
  unsigned short* Wot = (unsigned short*)ws; ws += (size_t)DD * DD * 2;      // [1024][1024]
  unsigned short* Qb  = (unsigned short*)ws; ws += (size_t)BB * LL * DD * 2;
  unsigned short* Kb  = (unsigned short*)ws; ws += (size_t)BB * LL * DD * 2;
  unsigned short* Vt  = (unsigned short*)ws; ws += (size_t)BB * HH * HD * LL * 2;
  unsigned short* Ctx = (unsigned short*)ws; ws += (size_t)BB * LL * DD * 2;

  int n4 = BB * LL * DD / 4;
  cvt_f32_bf16<<<(n4 + 255) / 256, 256, 0, stream>>>(x, Xbf, n4);
  cvt_transpose<<<dim3(3 * DD / 32, DD / 32), 256, 0, stream>>>(Wqkv, Wqt, DD, 3 * DD);
  cvt_transpose<<<dim3(DD / 32, DD / 32), 256, 0, stream>>>(Wout, Wot, DD, DD);

  qkv_gemm<<<dim3(3 * DD / 64, BB * LL / 128), 256, 0, stream>>>(Xbf, Wqt, bqkv, Qb, Kb, Vt);
  attn_fused<<<dim3(LL / 64, BB * HH), 256, 0, stream>>>(Qb, Kb, Vt, mask, attn, Ctx);
  out_gemm<<<dim3(DD / 64, BB * LL / 128), 256, 0, stream>>>(Ctx, Wot, bout, out);
}
